// QKVAttentionLegacy_20564303413998
// MI455X (gfx1250) — matrix operand order repaired and checked by reading, not compiled, not run on hardware
//
#include <hip/hip_runtime.h>

// ---------------------------------------------------------------------------
// Flash-attention for QKVAttentionLegacy on gfx1250 (MI455X, wave32, WMMA).
//   qkv: [4, 1536, 2048] fp32  ->  32 heads x {Q,K,V}[64, 2048] (t contiguous)
//   out: [4, 512, 2048] fp32   ==  [32, 64, 2048] flat
// f16 WMMA (16x16x32) for QK^T, PV, and row-sums (ones trick); f32 online
// softmax with DPP16 butterfly max-reduction; exp2-domain logits.
// ---------------------------------------------------------------------------

typedef __attribute__((ext_vector_type(16))) _Float16 v16h;
typedef __attribute__((ext_vector_type(8)))  _Float16 v8h;
typedef __attribute__((ext_vector_type(4)))  _Float16 v4h;
typedef __attribute__((ext_vector_type(8)))  float    v8f;

#define WMMA_F16(A, B, Cacc) \
  __builtin_amdgcn_wmma_f32_16x16x32_f16(false, (A), false, (B), (short)0, (Cacc), false, false)

// 16-lane butterfly max via DPP16 (stays inside each 16-lane row, zero LDS).
#define DPP_MAXF(v, ctrl) \
  (v) = fmaxf((v), __int_as_float(__builtin_amdgcn_mov_dpp( \
      __float_as_int(v), (ctrl), 0xF, 0xF, true)))

namespace {
constexpr int CH    = 64;    // head channel dim
constexpr int T     = 2048;  // sequence length
constexpr int BH    = 32;    // batch * heads
constexpr int SBLK  = 64;    // keys per block iteration
constexpr int LDSW  = 72;    // padded f16 row stride (144B, 16B-aligned)
constexpr int OSTR  = 20;    // padded f32 output row stride (80B, 16B-aligned)
constexpr int NW    = 8;     // waves per workgroup
constexpr int TBLK  = NW * 16;
// smem carve-up (bytes)
constexpr int KT_OFF = 0;                         // [64][72] f16 = 9216
constexpr int V_OFF  = 9216;                      // [64][72] f16 = 9216
constexpr int P_OFF  = 18432;                     // [8][16][72] f16 = 18432
constexpr int SMEM_B = 8 * 64 * OSTR * 4;         // out staging 40960 (superset)
constexpr float QSCALE = 0.125f * 1.44269504088896340736f; // 1/sqrt(64) * log2(e)
}

__global__ __launch_bounds__(256)
void qkv_attn_wmma_kernel(const float* __restrict__ qkv, float* __restrict__ out) {
  __shared__ alignas(16) unsigned char smem[SMEM_B];
  _Float16* __restrict__ sKt = (_Float16*)(smem + KT_OFF); // [s_local][c]
  _Float16* __restrict__ sV  = (_Float16*)(smem + V_OFF);  // [c][s_local]
  _Float16* __restrict__ sP  = (_Float16*)(smem + P_OFF);  // [wave][t_l][s_l]

  const int tid  = threadIdx.x;
  const int lane = tid & 31;
  const int w    = tid >> 5;          // wave 0..7
  const int l16  = lane & 15;
  const int half = lane >> 4;         // 0: lanes 0-15, 1: lanes 16-31
  const int a8   = half * 8;          // C-layout row offset

  const int head = blockIdx.y;
  const int t0   = blockIdx.x * TBLK + w * 16;

  const float* __restrict__ Qb = qkv + (size_t)head * 3 * CH * T;
  const float* __restrict__ Kb = Qb + (size_t)CH * T;
  const float* __restrict__ Vb = Kb + (size_t)CH * T;

  // ---- Q tile (16 rows) into WMMA A layout, scaled into exp2 domain ----
  // A 16x32 f16: lane holds M=lane%16; halves 0-15 -> K=a..a+15,
  // a = (lane<16 ? 0 : 16).
  v16h aq[2];
  {
    const int t = t0 + l16;
#pragma unroll
    for (int ch = 0; ch < 2; ++ch) {
      v16h q;
#pragma unroll
      for (int h = 0; h < 16; ++h) {
        const int k = half * 16 + h;
        const int c = ch * 32 + k;
        q[h] = (_Float16)(Qb[(size_t)c * T + t] * QSCALE);
      }
      aq[ch] = q;
    }
  }

  v16h bones;                 // all-ones B matrix for WMMA row-sums
#pragma unroll
  for (int h = 0; h < 16; ++h) bones[h] = (_Float16)1.0f;

  v8f o[4] = {};              // output tiles (C layout), cols n*16..n*16+15
  v8f osum = {};              // row-sum tile (every column == row sum)
  float mi[8];                // running row max (log2 domain), row = a8 + r
#pragma unroll
  for (int r = 0; r < 8; ++r) mi[r] = -3.0e38f;

  for (int sb = 0; sb < T; sb += SBLK) {
    __syncthreads();
    // ---- Stage K^T and V blocks (f32 -> f16) with b128 global loads ----
#pragma unroll
    for (int it = 0; it < 4; ++it) {
      const int j  = tid + it * 256;        // 1024 float4 tasks each for K, V
      const int c  = j >> 4;
      const int s4 = (j & 15) * 4;
      const float4 kv = *(const float4*)(Kb + (size_t)c * T + sb + s4);
      const float4 vv = *(const float4*)(Vb + (size_t)c * T + sb + s4);
      sKt[(s4 + 0) * LDSW + c] = (_Float16)kv.x;   // transpose: 4 b16 stores
      sKt[(s4 + 1) * LDSW + c] = (_Float16)kv.y;
      sKt[(s4 + 2) * LDSW + c] = (_Float16)kv.z;
      sKt[(s4 + 3) * LDSW + c] = (_Float16)kv.w;
      const v4h vh = { (_Float16)vv.x, (_Float16)vv.y,
                       (_Float16)vv.z, (_Float16)vv.w };
      *(v4h*)(sV + c * LDSW + s4) = vh;            // natural: one b64 store
    }
    if (sb + SBLK < T && tid < CH) {   // pull next K/V block toward caches
      __builtin_prefetch(Kb + (size_t)tid * T + sb + SBLK, 0, 0);
      __builtin_prefetch(Vb + (size_t)tid * T + sb + SBLK, 0, 0);
    }
    __syncthreads();

    // ---- S = (Q*s)^T (K*s) * log2(e) for four 16-wide key subtiles ----
    v8f st[4];
#pragma unroll
    for (int sub = 0; sub < 4; ++sub) {
      const _Float16* krow = sKt + (sub * 16 + l16) * LDSW;
      v8f acc = {};
#pragma unroll
      for (int ch = 0; ch < 2; ++ch) {
        const _Float16* p = krow + ch * 32 + half * 16;  // B: K=c consecutive
        const v8h blo = *(const v8h*)(p);
        const v8h bhi = *(const v8h*)(p + 8);
        const v16h b = __builtin_shufflevector(blo, bhi,
            0,1,2,3,4,5,6,7,8,9,10,11,12,13,14,15);
        acc = WMMA_F16(aq[ch], b, acc);
      }
      st[sub] = acc;
    }

    // ---- Online softmax (base-2): max via DPP butterfly, sums via WMMA ----
#pragma unroll
    for (int r = 0; r < 8; ++r) {
      float v = fmaxf(fmaxf(st[0][r], st[1][r]), fmaxf(st[2][r], st[3][r]));
      DPP_MAXF(v, 0xB1);   // quad_perm xor1
      DPP_MAXF(v, 0x4E);   // quad_perm xor2
      DPP_MAXF(v, 0x141);  // row_half_mirror (xor7)
      DPP_MAXF(v, 0x140);  // row_mirror (xor15)
      const float mnew  = fmaxf(mi[r], v);
      const float alpha = exp2f(mi[r] - mnew);
      mi[r] = mnew;

#pragma unroll
      for (int n = 0; n < 4; ++n) o[n][r] *= alpha;  // rescale accumulators
      osum[r] *= alpha;

      // p = 2^(s - m); spill f16 P: row t_local = a8+r, col = k*16+l16
#pragma unroll
      for (int k = 0; k < 4; ++k) {
        const float pv = exp2f(st[k][r] - mnew);
        sP[(w * 16 + a8 + r) * LDSW + k * 16 + l16] = (_Float16)pv;
      }
    }

    // ---- O += P*V ; row sums += P*ones  (A = P from LDS, B = V from LDS) ----
#pragma unroll
    for (int ch = 0; ch < 2; ++ch) {
      const _Float16* prow = sP + (w * 16 + l16) * LDSW + ch * 32;
      const v8h alo = *(const v8h*)(prow + half * 16);        // halves 0-7
      const v8h ahi = *(const v8h*)(prow + half * 16 + 8);   // halves 8-15
      const v16h A = __builtin_shufflevector(alo, ahi,
          0,1,2,3,4,5,6,7,8,9,10,11,12,13,14,15);
#pragma unroll
      for (int n = 0; n < 4; ++n) {
        const _Float16* vrow = sV + (n * 16 + l16) * LDSW + ch * 32 + half * 16;
        const v8h blo = *(const v8h*)(vrow);
        const v8h bhi = *(const v8h*)(vrow + 8);
        const v16h B = __builtin_shufflevector(blo, bhi,
            0,1,2,3,4,5,6,7,8,9,10,11,12,13,14,15);
        o[n] = WMMA_F16(A, B, o[n]);
      }
      osum = WMMA_F16(A, bones, osum);   // every column = row sum
    }
  }

  // ---- Normalize, transpose via LDS, coalesced b128 stores ----
  __syncthreads();                       // smem reused for output staging
  float* __restrict__ sO = (float*)smem + w * CH * OSTR;  // [c][t_local]
#pragma unroll
  for (int n = 0; n < 4; ++n)
#pragma unroll
    for (int r = 0; r < 8; ++r)
      sO[(n * 16 + l16) * OSTR + a8 + r] = o[n][r] / osum[r];

  // wave-local: DS ops are in-order within a wave, no barrier needed
#pragma unroll
  for (int i = 0; i < 8; ++i) {
    const int idx = lane + 32 * i;       // 256 tasks: 64 rows x 4 quads
    const int c   = idx >> 2;
    const int q   = (idx & 3) * 4;
    const float4 val = *(const float4*)(sO + c * OSTR + q);
    *(float4*)(out + ((size_t)head * CH + c) * T + t0 + q) = val;
  }
}

extern "C" void kernel_launch(void* const* d_in, const int* in_sizes, int n_in,
                              void* d_out, int out_size, void* d_ws, size_t ws_size,
                              hipStream_t stream) {
  (void)in_sizes; (void)n_in; (void)out_size; (void)d_ws; (void)ws_size;
  const float* qkv = (const float*)d_in[0];
  float* out = (float*)d_out;
  dim3 grid(T / TBLK, BH);   // 16 query tiles of 128 rows  x  32 heads
  qkv_attn_wmma_kernel<<<grid, 256, 0, stream>>>(qkv, out);
}
